// BitLinear1b_88905823027951
// MI455X (gfx1250) — compile-verified, matrix operand order checked by reading, not run on hardware
//
#include <hip/hip_runtime.h>
#include <hip/hip_bf16.h>
#include <stdint.h>

typedef int   v8i __attribute__((ext_vector_type(8)));
typedef float v8f __attribute__((ext_vector_type(8)));

#define QGROUP 64
#define QMAXF  127.0f

// Fixed problem geometry (reference: D_IN = D_OUT = 4096, group 64)
constexpr int KDIM     = 4096;
constexpr int NGROUPS  = KDIM / QGROUP;          // 64
constexpr int BLKBYTES = 32 * 32;                // 16-row x 64-K fragment block: 32 lanes x 32B
constexpr int GBYTES   = NGROUPS * BLKBYTES;     // bytes per 16-row block across all groups = 64 KB

// Swizzled operand layout:
//   buf[ ((row16_block * NGROUPS + g) * 32 + lane) * 32 + byte ]
// lane = hi*16 + (row & 15); lane's 32 bytes = K-chunks {hi*8, 16+hi*8, 32+hi*8, 48+hi*8}
// (exactly the 8-bit WMMA A/B fragment register image -> contiguous b128 loads)

// ---------------------------------------------------------------------------
// Kernel 1: per-token group-of-64 absmax quantize, writing swizzled int8.
// ---------------------------------------------------------------------------
__global__ __launch_bounds__(256)
void quantize_act_kernel(const float* __restrict__ x,
                         int8_t* __restrict__ q,        // swizzled [M/16][G][32][32]
                         float* __restrict__ scale_t,   // [G][M]
                         int M) {
    const int lane = threadIdx.x & 31;
    const int wave = threadIdx.x >> 5;
    const int gid  = blockIdx.x * (blockDim.x >> 5) + wave;
    const int m    = gid / NGROUPS;
    const int g    = gid - m * NGROUPS;
    if (m >= M) return;

    const size_t base = (size_t)m * KDIM + (size_t)g * QGROUP;
    float2 v = *(const float2*)(x + base + lane * 2);

    float a = fmaxf(fabsf(v.x), fabsf(v.y));
#pragma unroll
    for (int off = 16; off > 0; off >>= 1)
        a = fmaxf(a, __shfl_xor(a, off, 32));

    const float scale = fmaxf(a, 1e-8f);
    const float inv   = QMAXF / scale;
    float q0 = fminf(fmaxf(rintf(v.x * inv), -QMAXF), QMAXF);
    float q1 = fminf(fmaxf(rintf(v.y * inv), -QMAXF), QMAXF);
    char2 qp;
    qp.x = (int8_t)(int)q0;
    qp.y = (int8_t)(int)q1;

    const int k0    = lane * 2;
    const int chunk = k0 >> 3;
    const int dlane = (chunk & 1) * 16 + (m & 15);
    const int pos   = ((chunk >> 1) << 3) + (k0 & 7);
    int8_t* dst = q + ((((size_t)(m >> 4) * NGROUPS + g) * 32) + dlane) * 32 + pos;
    *(char2*)dst = qp;

    if (lane == 0)
        scale_t[(size_t)g * M + m] = scale * (1.0f / QMAXF);  // fold 1/127
}

// ---------------------------------------------------------------------------
// Kernel 2: per-row centered binarization of W, writing swizzled int8 signs.
// ---------------------------------------------------------------------------
__global__ __launch_bounds__(256)
void binarize_w_kernel(const float* __restrict__ W,
                       int8_t* __restrict__ wsgn,   // swizzled [N/16][G][32][32]
                       float* __restrict__ alpha,
                       int N) {
    __shared__ float red[256];
    const int o = blockIdx.x;
    const int t = threadIdx.x;
    const float4* row4 = (const float4*)(W + (size_t)o * KDIM);
    const int nvec = KDIM / 4;

    float s = 0.0f;
    for (int i = t; i < nvec; i += blockDim.x) {
        float4 v = row4[i];
        s += v.x + v.y + v.z + v.w;
    }
    red[t] = s;
    __syncthreads();
    for (int st = 128; st > 0; st >>= 1) {
        if (t < st) red[t] += red[t + st];
        __syncthreads();
    }
    const float mean = red[0] / (float)KDIM;
    __syncthreads();

    s = 0.0f;
    for (int i = t; i < nvec; i += blockDim.x) {
        float4 v = row4[i];
        s += fabsf(v.x - mean) + fabsf(v.y - mean) +
             fabsf(v.z - mean) + fabsf(v.w - mean);
    }
    red[t] = s;
    __syncthreads();
    for (int st = 128; st > 0; st >>= 1) {
        if (t < st) red[t] += red[t + st];
        __syncthreads();
    }
    if (t == 0) alpha[o] = red[0] / (float)KDIM;

    const size_t nblk = (size_t)(o >> 4) * NGROUPS;
    const int    lrn  = o & 15;
    for (int i = t; i < nvec; i += blockDim.x) {
        float4 v = row4[i];
        char4 sg;
        float c;
        c = v.x - mean; sg.x = (c > 0.0f) ? 1 : ((c < 0.0f) ? -1 : 0);
        c = v.y - mean; sg.y = (c > 0.0f) ? 1 : ((c < 0.0f) ? -1 : 0);
        c = v.z - mean; sg.z = (c > 0.0f) ? 1 : ((c < 0.0f) ? -1 : 0);
        c = v.w - mean; sg.w = (c > 0.0f) ? 1 : ((c < 0.0f) ? -1 : 0);

        const int k     = i * 4;
        const int g     = k >> 6;
        const int kk    = k & 63;
        const int chunk = kk >> 3;
        const int dlane = (chunk & 1) * 16 + lrn;
        const int pos   = ((chunk >> 1) << 3) + (kk & 7);
        int8_t* dst = wsgn + (((nblk + g) * 32) + dlane) * 32 + pos;
        *(char4*)dst = sg;
    }
}

// ---------------------------------------------------------------------------
__device__ __forceinline__ v8i load_frag_sw(const int8_t* __restrict__ p) {
    int4 a = ((const int4*)p)[0];
    int4 b = ((const int4*)p)[1];
    v8i r;
    r[0] = a.x; r[1] = a.y; r[2] = a.z; r[3] = a.w;
    r[4] = b.x; r[5] = b.y; r[6] = b.z; r[7] = b.w;
    return r;
}

__device__ __forceinline__ void load_group(const int8_t* __restrict__ pA,
                                           const int8_t* __restrict__ pB,
                                           const float* __restrict__ sp,
                                           v8i Af[2], v8i Bf[4], v8f s[2]) {
#pragma unroll
    for (int t = 0; t < 2; ++t)
        Af[t] = load_frag_sw(pA + t * GBYTES);     // immediate offsets 0, 64K
#pragma unroll
    for (int u = 0; u < 4; ++u)
        Bf[u] = load_frag_sw(pB + u * GBYTES);     // 0, 64K, 128K, 192K
#pragma unroll
    for (int t = 0; t < 2; ++t) {
        float4 f0 = ((const float4*)(sp + t * 16))[0];
        float4 f1 = ((const float4*)(sp + t * 16))[1];
        s[t][0] = f0.x; s[t][1] = f0.y; s[t][2] = f0.z; s[t][3] = f0.w;
        s[t][4] = f1.x; s[t][5] = f1.y; s[t][6] = f1.z; s[t][7] = f1.w;
    }
}

__device__ __forceinline__ void issue_wmma(const v8i Af[2], const v8i Bf[4],
                                           v8i ci[2][4]) {
    const v8i izero = {0, 0, 0, 0, 0, 0, 0, 0};
#pragma unroll
    for (int t = 0; t < 2; ++t)
#pragma unroll
        for (int u = 0; u < 4; ++u)
            ci[t][u] = __builtin_amdgcn_wmma_i32_16x16x64_iu8(
                true, Af[t], true, Bf[u], izero, false, false);
}

// Issue this group's WMMA per tile, consume LAST group's result for the same
// tile, then rotate the buffer. Temp's live range ends at ci's redefinition,
// so no copies; only ~9 result tiles are ever live.
__device__ __forceinline__ void step_group(const v8i Af[2], const v8i Bf[4],
                                           const v8f s[2],
                                           v8i ci[2][4], v8f acc[2][4]) {
    const v8i izero = {0, 0, 0, 0, 0, 0, 0, 0};
#pragma unroll
    for (int t = 0; t < 2; ++t)
#pragma unroll
        for (int u = 0; u < 4; ++u) {
            v8i nw = __builtin_amdgcn_wmma_i32_16x16x64_iu8(
                true, Af[t], true, Bf[u], izero, false, false);
#pragma unroll
            for (int r = 0; r < 8; ++r)
                acc[t][u][r] += s[t][r] * (float)ci[t][u][r];
            ci[t][u] = nw;
        }
}

__device__ __forceinline__ void consume_group(const v8i ci[2][4],
                                              const v8f s[2],
                                              v8f acc[2][4]) {
#pragma unroll
    for (int t = 0; t < 2; ++t)
#pragma unroll
        for (int u = 0; u < 4; ++u)
#pragma unroll
            for (int r = 0; r < 8; ++r)
                acc[t][u][r] += s[t][r] * (float)ci[t][u][r];
}

// ---------------------------------------------------------------------------
// Kernel 3: GEMM  y[m,n] = alpha[n] * sum_g scale_t[g,m] * (q[m,g] . s[n,g])
// Block = 256 threads = 8 wave32, tile 128(M) x 128(N); wave tile 32x64.
// Rotating single result buffer, consumption lagged one group behind issue.
// ---------------------------------------------------------------------------
__global__ __launch_bounds__(256)
void bitlinear_wmma_kernel(const int8_t* __restrict__ qx,
                           const int8_t* __restrict__ wsgn,
                           const float*  __restrict__ scale_t,
                           const float*  __restrict__ alpha,
                           float* __restrict__ y,
                           int M, int N) {
    const int lane = threadIdx.x & 31;
    const int wave = threadIdx.x >> 5;
    const int wm   = wave >> 1;          // 0..3
    const int wn   = wave & 1;           // 0..1
    const int lr   = lane & 15;
    const int hi   = lane >> 4;

    const int m0 = blockIdx.y * 128 + wm * 32;
    const int n0 = blockIdx.x * 128 + wn * 64;

    v8f acc[2][4];
#pragma unroll
    for (int t = 0; t < 2; ++t)
#pragma unroll
        for (int u = 0; u < 4; ++u)
#pragma unroll
            for (int r = 0; r < 8; ++r)
                acc[t][u][r] = 0.0f;

    const int8_t* pA = qx   + (size_t)(m0 >> 4) * GBYTES + lane * 32;
    const int8_t* pB = wsgn + (size_t)(n0 >> 4) * GBYTES + lane * 32;
    const float*  sp = scale_t + m0 + hi * 8;

    v8i Af[2], Bf[4];
    v8i ci[2][4];
    v8f sc[2], sn[2];

    // ---- prologue: group 0 resident in ci, its scales in sc ----
    load_group(pA, pB, sp, Af, Bf, sc);
    issue_wmma(Af, Bf, ci);
    pA += BLKBYTES; pB += BLKBYTES; sp += M;

    int g = 1;
    for (; g + 1 < NGROUPS; g += 2) {
        __builtin_prefetch(pA + BLKBYTES, 0, 1);   // global_prefetch_b8
        __builtin_prefetch(pB + BLKBYTES, 0, 1);

        load_group(pA, pB, sp, Af, Bf, sn);   // group g (scales -> sn)
        step_group(Af, Bf, sc, ci, acc);      // consume g-1 (sc), ci <- g
        pA += BLKBYTES; pB += BLKBYTES; sp += M;

        load_group(pA, pB, sp, Af, Bf, sc);   // group g+1 (scales -> sc)
        step_group(Af, Bf, sn, ci, acc);      // consume g (sn), ci <- g+1
        pA += BLKBYTES; pB += BLKBYTES; sp += M;
    }

    // ---- tail: exactly one group left (NGROUPS even) ----
    load_group(pA, pB, sp, Af, Bf, sn);       // group 63
    step_group(Af, Bf, sc, ci, acc);          // consume 62, ci <- 63
    consume_group(ci, sn, acc);               // consume 63

    // ---- store: y[m,n] = acc * alpha[n] ----
#pragma unroll
    for (int u = 0; u < 4; ++u) {
        const int n = n0 + u * 16 + lr;
        const float av = alpha[n];
#pragma unroll
        for (int t = 0; t < 2; ++t) {
#pragma unroll
            for (int r = 0; r < 8; ++r) {
                const int mm = m0 + t * 16 + hi * 8 + r;
                y[(size_t)mm * N + n] = acc[t][u][r] * av;
            }
        }
    }
}

// ---------------------------------------------------------------------------
extern "C" void kernel_launch(void* const* d_in, const int* in_sizes, int n_in,
                              void* d_out, int out_size, void* d_ws, size_t ws_size,
                              hipStream_t stream) {
    const float* x = (const float*)d_in[0];   // [M, K]
    const float* W = (const float*)d_in[1];   // [N, K]
    const int M = in_sizes[0] / KDIM;         // 8192
    const int N = in_sizes[1] / KDIM;         // 4096

    uint8_t* ws = (uint8_t*)d_ws;
    size_t off = 0;
    auto carve = [&](size_t bytes) -> void* {
        void* p = ws + off;
        off = (off + bytes + 255) & ~(size_t)255;
        return p;
    };
    int8_t* qx      = (int8_t*)carve((size_t)M * KDIM);                    // 32 MB
    float*  scale_t = (float*) carve((size_t)NGROUPS * M * sizeof(float)); //  2 MB
    int8_t* wsgn    = (int8_t*)carve((size_t)N * KDIM);                    // 16 MB
    float*  alpha   = (float*) carve((size_t)N * sizeof(float));
    (void)ws_size; (void)n_in; (void)out_size;

    const int groups = M * NGROUPS;
    quantize_act_kernel<<<(groups + 7) / 8, 256, 0, stream>>>(x, qx, scale_t, M);
    binarize_w_kernel<<<N, 256, 0, stream>>>(W, wsgn, alpha, N);

    dim3 grid(N / 128, M / 128);
    bitlinear_wmma_kernel<<<grid, 256, 0, stream>>>(qx, wsgn, scale_t, alpha,
                                                    (float*)d_out, M, N);
}
